// MLAEncoderSelfAttention_73787538145764
// MI455X (gfx1250) — matrix operand order repaired and checked by reading, not compiled, not run on hardware
//
#include <hip/hip_runtime.h>

typedef __bf16 bf16;
typedef __attribute__((ext_vector_type(16))) __bf16 v16bf;
typedef __attribute__((ext_vector_type(8)))  float  v8f;

#define B_    2
#define S_    2048
#define E_    1024
#define H_    16
#define HD_   64
#define NOPE_ 32
#define ROPE_ 32
#define QL_   512
#define KVL_  512

// ---------- bf16 helpers (explicit round-to-nearest-even) ----------
__device__ __host__ inline bf16 f2bf(float x) {
  union { float f; unsigned u; } v; v.f = x;
  unsigned r = v.u + 0x7FFFu + ((v.u >> 16) & 1u);
  unsigned short h = (unsigned short)(r >> 16);
  return __builtin_bit_cast(bf16, h);
}
__device__ __host__ inline float bf2f(bf16 x) {
  unsigned short h = __builtin_bit_cast(unsigned short, x);
  union { unsigned u; float f; } v; v.u = ((unsigned)h) << 16;
  return v.f;
}

// ---------- elementwise f32 -> bf16 ----------
__global__ void k_cvt_f32_bf16(const float* __restrict__ in, bf16* __restrict__ out, long n) {
  long i = (long)blockIdx.x * blockDim.x + threadIdx.x;
  if (i < n) out[i] = f2bf(in[i]);
}

// ---------- generic GEMM: C[M,N] (f32) = A[M,K] (bf16, row-major) x Bw[N,K]^T (bf16, row-major) ----------
// one wave = one 16x32 C tile (A fragment reused across 2 B subtiles); WG = 8 waves = 64x64 tile.
// Requires M%64==0, N%32==0, K%32==0.
__global__ __launch_bounds__(256)
void k_gemm_bf16(const bf16* __restrict__ A, const bf16* __restrict__ Bw,
                 float* __restrict__ C, int M, int N, int K) {
  const int lane = threadIdx.x & 31;
  const int wave = threadIdx.x >> 5;
  const int wm = wave >> 1, wn = wave & 1;
  const int mbase = blockIdx.y * 64 + wm * 16;
  const int nbase = blockIdx.x * 64 + wn * 32;
  if (mbase >= M || nbase >= N) return;     // wave-uniform: EXEC stays all-ones
  const int half = lane >> 4;
  const int ln16 = lane & 15;

  union { v16bf v; uint4 q[2]; } af, b0, b1;
  v8f acc0 = {}, acc1 = {};
  const bf16* arow  = A  + (size_t)(mbase + ln16) * K;
  const bf16* brow0 = Bw + (size_t)(nbase + ln16) * K;
  const bf16* brow1 = Bw + (size_t)(nbase + 16 + ln16) * K;

  for (int k0 = 0; k0 < K; k0 += 32) {
    af.q[0] = *(const uint4*)(arow + k0 + half * 16);
    af.q[1] = *(const uint4*)(arow + k0 + half * 16 + 8);
    const bf16* bp0 = brow0 + k0 + half * 16;
    const bf16* bp1 = brow1 + k0 + half * 16;
    b0.q[0] = *(const uint4*)(bp0);
    b0.q[1] = *(const uint4*)(bp0 + 8);
    b1.q[0] = *(const uint4*)(bp1);
    b1.q[1] = *(const uint4*)(bp1 + 8);
    acc0 = __builtin_amdgcn_wmma_f32_16x16x32_bf16(false, af.v, false, b0.v,
                                                   (short)0, acc0, false, false);
    acc1 = __builtin_amdgcn_wmma_f32_16x16x32_bf16(false, af.v, false, b1.v,
                                                   (short)0, acc1, false, false);
  }
  float* crow = C + (size_t)(mbase + half * 8) * N + nbase + ln16;
#pragma unroll
  for (int r = 0; r < 8; ++r) {
    crow[(size_t)r * N]      = acc0[r];
    crow[(size_t)r * N + 16] = acc1[r];
  }
}

// ---------- layernorm over 512 cols (one block per row), f32 in -> bf16 out ----------
__global__ __launch_bounds__(256)
void k_layernorm_bf16(const float* __restrict__ in, int in_stride,
                      const float* __restrict__ g, const float* __restrict__ b,
                      bf16* __restrict__ out) {
  const int row = blockIdx.x, tid = threadIdx.x;
  const float* x = in + (size_t)row * in_stride;
  __shared__ float red[256];
  float v0 = x[tid], v1 = x[tid + 256];
  red[tid] = v0 + v1; __syncthreads();
  for (int off = 128; off > 0; off >>= 1) {
    if (tid < off) red[tid] += red[tid + off];
    __syncthreads();
  }
  float mu = red[0] * (1.0f / 512.0f);
  __syncthreads();
  float d0 = v0 - mu, d1 = v1 - mu;
  red[tid] = d0 * d0 + d1 * d1; __syncthreads();
  for (int off = 128; off > 0; off >>= 1) {
    if (tid < off) red[tid] += red[tid + off];
    __syncthreads();
  }
  float rs = rsqrtf(red[0] * (1.0f / 512.0f) + 1e-5f);
  bf16* o = out + (size_t)row * 512;
  o[tid]       = f2bf(d0 * rs * g[tid]       + b[tid]);
  o[tid + 256] = f2bf(d1 * rs * g[tid + 256] + b[tid + 256]);
}

// ---------- pack q: (B,S,H,HD) f32 -> q_full (B,H,S,HD) bf16 with RoPE on last 32 dims ----------
__global__ void k_pack_q(const float* __restrict__ q, bf16* __restrict__ qfull) {
  long i = (long)blockIdx.x * 256 + threadIdx.x;   // over B*H*S*64
  int d = (int)(i & 63);
  long t = i >> 6;
  int s = (int)(t % S_); long t2 = t / S_;
  int h = (int)(t2 % H_); int b = (int)(t2 / H_);
  const float* src = q + (size_t)(b * S_ + s) * (H_ * HD_) + h * HD_;
  float val;
  if (d < NOPE_) {
    val = src[d];
  } else {
    int j = d - NOPE_, i2 = j >> 1;
    float x1 = src[NOPE_ + 2 * i2], x2 = src[NOPE_ + 2 * i2 + 1];
    float f = (float)s * powf(10000.0f, -(float)i2 / 16.0f);
    float c = cosf(f), sn = sinf(f);
    val = (j & 1) ? (x1 * sn + x2 * c) : (x1 * c - x2 * sn);
  }
  qfull[i] = f2bf(val);
}

// ---------- pack k_full (B,H,S,HD) and V TRANSPOSED (B,H,HD,S), both bf16 ----------
__global__ void k_pack_kv(const float* __restrict__ kvd, const float* __restrict__ kva,
                          bf16* __restrict__ kfull, bf16* __restrict__ vT) {
  long i = (long)blockIdx.x * 256 + threadIdx.x;   // over B*H*S*64
  int d = (int)(i & 63);
  long t = i >> 6;
  int s = (int)(t % S_); long t2 = t / S_;
  int h = (int)(t2 % H_); int b = (int)(t2 / H_);
  const int bh = b * H_ + h;
  const float* kvrow = kvd + (size_t)(b * S_ + s) * (H_ * (NOPE_ + HD_)) + h * (NOPE_ + HD_);
  // V transposed: vT[bh][d][s]
  vT[((size_t)bh * HD_ + d) * S_ + s] = f2bf(kvrow[NOPE_ + d]);
  float val;
  if (d < NOPE_) {
    val = kvrow[d];
  } else {
    int j = d - NOPE_, i2 = j >> 1;
    const float* pe = kva + (size_t)(b * S_ + s) * (KVL_ + ROPE_) + KVL_;
    float x1 = pe[2 * i2], x2 = pe[2 * i2 + 1];
    float f = (float)s * powf(10000.0f, -(float)i2 / 16.0f);
    float c = cosf(f), sn = sinf(f);
    val = (j & 1) ? (x1 * sn + x2 * c) : (x1 * c - x2 * sn);
  }
  kfull[i] = f2bf(val);
}

// ---------- flash attention: one wave per 16 query rows, 8 waves/WG ----------
__global__ __launch_bounds__(256)
void k_attention(const bf16* __restrict__ qfull, const bf16* __restrict__ kfull,
                 const bf16* __restrict__ vT, float* __restrict__ outp) {
  __shared__ float sp[8][16][32];                 // per-wave 16x32 P staging
  const int lane = threadIdx.x & 31, wave = threadIdx.x >> 5;
  const int half = lane >> 4, ln16 = lane & 15;
  const int nqb  = S_ / 128;
  const int bh   = blockIdx.x / nqb;
  const int qblk = blockIdx.x % nqb;
  const int qbase = qblk * 128 + wave * 16;
  const int b = bh / H_, h = bh % H_;
  const bf16* qf = qfull + (size_t)bh * S_ * HD_;
  const bf16* kf = kfull + (size_t)bh * S_ * HD_;
  const bf16* vf = vT    + (size_t)bh * HD_ * S_;   // [d][s]

  union F { v16bf v; uint4 q[2]; bf16 e[16]; };
  F aq0, aq1;
  {
    const bf16* qr = qf + (size_t)(qbase + ln16) * HD_;
    aq0.q[0] = *(const uint4*)(qr +  0 + half * 16);
    aq0.q[1] = *(const uint4*)(qr +  8 + half * 16);
    aq1.q[0] = *(const uint4*)(qr + 32 + half * 16);
    aq1.q[1] = *(const uint4*)(qr + 40 + half * 16);
  }
  v8f O0 = {}, O1 = {}, O2 = {}, O3 = {};
  float rowM[8], rowL[8];
#pragma unroll
  for (int r = 0; r < 8; ++r) { rowM[r] = -3.0e38f; rowL[r] = 0.0f; }
  const float scale = 0.125f;                     // 1/sqrt(64)
  float (*pS)[32] = sp[wave];

  // per-lane V base pointers for the four 16-col tiles (d = t*16 + ln16, keys contiguous)
  const bf16* vp0 = vf + (size_t)(0 * 16 + ln16) * S_ + half * 16;
  const bf16* vp1 = vf + (size_t)(1 * 16 + ln16) * S_ + half * 16;
  const bf16* vp2 = vf + (size_t)(2 * 16 + ln16) * S_ + half * 16;
  const bf16* vp3 = vf + (size_t)(3 * 16 + ln16) * S_ + half * 16;

  for (int kb = 0; kb < S_; kb += 32) {
    F bk;
    v8f s0 = {}, s1 = {};
    const bf16* kr0 = kf + (size_t)(kb + ln16) * HD_;
    const bf16* kr1 = kf + (size_t)(kb + 16 + ln16) * HD_;
    // prefetch next key block (speculative; dropped if invalid)
    __builtin_prefetch(kr0 + 32 * HD_, 0, 1);
    __builtin_prefetch(vp0 + kb + 32, 0, 1);
    // scores, d-chunk 0
    bk.q[0] = *(const uint4*)(kr0 + half * 16);
    bk.q[1] = *(const uint4*)(kr0 + half * 16 + 8);
    s0 = __builtin_amdgcn_wmma_f32_16x16x32_bf16(false, aq0.v, false, bk.v, (short)0, s0, false, false);
    bk.q[0] = *(const uint4*)(kr1 + half * 16);
    bk.q[1] = *(const uint4*)(kr1 + half * 16 + 8);
    s1 = __builtin_amdgcn_wmma_f32_16x16x32_bf16(false, aq0.v, false, bk.v, (short)0, s1, false, false);
    // scores, d-chunk 1
    bk.q[0] = *(const uint4*)(kr0 + 32 + half * 16);
    bk.q[1] = *(const uint4*)(kr0 + 32 + half * 16 + 8);
    s0 = __builtin_amdgcn_wmma_f32_16x16x32_bf16(false, aq1.v, false, bk.v, (short)0, s0, false, false);
    bk.q[0] = *(const uint4*)(kr1 + 32 + half * 16);
    bk.q[1] = *(const uint4*)(kr1 + 32 + half * 16 + 8);
    s1 = __builtin_amdgcn_wmma_f32_16x16x32_bf16(false, aq1.v, false, bk.v, (short)0, s1, false, false);

    // online softmax update (row r lives across the 16 lanes of this half-group)
#pragma unroll
    for (int r = 0; r < 8; ++r) {
      float a0 = s0[r] * scale, a1 = s1[r] * scale;
      float mx = fmaxf(a0, a1);
      mx = fmaxf(mx, __shfl_xor(mx, 1));
      mx = fmaxf(mx, __shfl_xor(mx, 2));
      mx = fmaxf(mx, __shfl_xor(mx, 4));
      mx = fmaxf(mx, __shfl_xor(mx, 8));
      float newM = fmaxf(rowM[r], mx);
      float alpha = __expf(rowM[r] - newM);
      rowM[r] = newM;
      float p0 = __expf(a0 - newM), p1 = __expf(a1 - newM);
      float rs = p0 + p1;
      rs += __shfl_xor(rs, 1);
      rs += __shfl_xor(rs, 2);
      rs += __shfl_xor(rs, 4);
      rs += __shfl_xor(rs, 8);
      rowL[r] = rowL[r] * alpha + rs;
      O0[r] *= alpha; O1[r] *= alpha; O2[r] *= alpha; O3[r] *= alpha;
      int m = r + half * 8;
      pS[m][ln16]      = p0;
      pS[m][16 + ln16] = p1;
    }

    // relayout P: C-form -> A-form via wave-local LDS (DS ops in-order within wave)
    F pa;
#pragma unroll
    for (int e = 0; e < 8; ++e) pa.e[e]     = f2bf(pS[ln16][half * 16 + e]);
#pragma unroll
    for (int e = 0; e < 8; ++e) pa.e[8 + e] = f2bf(pS[ln16][half * 16 + 8 + e]);

    // O += P(16x32) x V(32x64): V^T makes each B fragment two contiguous b128 loads
    {
      F bv;
      bv.q[0] = *(const uint4*)(vp0 + kb);
      bv.q[1] = *(const uint4*)(vp0 + kb + 8);
      O0 = __builtin_amdgcn_wmma_f32_16x16x32_bf16(false, pa.v, false, bv.v, (short)0, O0, false, false);
      bv.q[0] = *(const uint4*)(vp1 + kb);
      bv.q[1] = *(const uint4*)(vp1 + kb + 8);
      O1 = __builtin_amdgcn_wmma_f32_16x16x32_bf16(false, pa.v, false, bv.v, (short)0, O1, false, false);
      bv.q[0] = *(const uint4*)(vp2 + kb);
      bv.q[1] = *(const uint4*)(vp2 + kb + 8);
      O2 = __builtin_amdgcn_wmma_f32_16x16x32_bf16(false, pa.v, false, bv.v, (short)0, O2, false, false);
      bv.q[0] = *(const uint4*)(vp3 + kb);
      bv.q[1] = *(const uint4*)(vp3 + kb + 8);
      O3 = __builtin_amdgcn_wmma_f32_16x16x32_bf16(false, pa.v, false, bv.v, (short)0, O3, false, false);
    }
  }

  // write normalized output in (B,S,H*HD) layout
#pragma unroll
  for (int r = 0; r < 8; ++r) {
    int srow = qbase + r + half * 8;
    float inv = 1.0f / rowL[r];
    float* dst = outp + (size_t)(b * S_ + srow) * (H_ * HD_) + h * HD_ + ln16;
    dst[0]  = O0[r] * inv;
    dst[16] = O1[r] * inv;
    dst[32] = O2[r] * inv;
    dst[48] = O3[r] * inv;
  }
}

// ---------------------------------------------------------------------------
extern "C" void kernel_launch(void* const* d_in, const int* in_sizes, int n_in,
                              void* d_out, int out_size, void* d_ws, size_t ws_size,
                              hipStream_t stream) {
  (void)in_sizes; (void)n_in; (void)out_size; (void)ws_size;
  const float* x     = (const float*)d_in[0];
  const float* Wqa   = (const float*)d_in[1];
  const float* g_qa  = (const float*)d_in[2];
  const float* b_qa  = (const float*)d_in[3];
  const float* Wqb   = (const float*)d_in[4];
  const float* Wkva  = (const float*)d_in[5];
  const float* g_kva = (const float*)d_in[6];
  const float* b_kva = (const float*)d_in[7];
  const float* Wkvb  = (const float*)d_in[8];
  const float* Wout  = (const float*)d_in[9];
  float* out = (float*)d_out;
  char* ws = (char*)d_ws;

  const long M = (long)B_ * S_;                   // 4096
  size_t off = 0;
  auto take = [&](size_t bytes) { size_t r = off; off += (bytes + 255) & ~(size_t)255; return r; };

  bf16*  xb     = (bf16*)(ws + take(M * E_ * 2));                 // reused later as attno_b
  bf16*  wqa_b  = (bf16*)(ws + take((size_t)QL_ * E_ * 2));
  bf16*  wqb_b  = (bf16*)(ws + take((size_t)(H_ * HD_) * QL_ * 2));
  bf16*  wkva_b = (bf16*)(ws + take((size_t)(KVL_ + ROPE_) * E_ * 2));
  bf16*  wkvb_b = (bf16*)(ws + take((size_t)(H_ * (NOPE_ + HD_)) * KVL_ * 2));
  bf16*  wout_b = (bf16*)(ws + take((size_t)E_ * (H_ * HD_) * 2));
  float* qc     = (float*)(ws + take(M * QL_ * 4));
  bf16*  qcln   = (bf16*)(ws + take(M * QL_ * 2));
  float* kva    = (float*)(ws + take(M * (KVL_ + ROPE_) * 4));
  bf16*  ckvln  = (bf16*)(ws + take(M * KVL_ * 2));
  float* qf32   = (float*)(ws + take(M * (H_ * HD_) * 4));        // reused later as attno f32
  float* kvd    = (float*)(ws + take(M * (H_ * (NOPE_ + HD_)) * 4));
  bf16*  qfull  = (bf16*)(ws + take((size_t)B_ * H_ * S_ * HD_ * 2));
  bf16*  kfull  = (bf16*)(ws + take((size_t)B_ * H_ * S_ * HD_ * 2));
  bf16*  vTb    = (bf16*)(ws + take((size_t)B_ * H_ * S_ * HD_ * 2));
  float* attno  = qf32;                                           // q_f32 dead after pack_q
  bf16*  attno_b = xb;                                            // x_bf16 dead after kva GEMM

  auto cvt = [&](const float* in, bf16* o, long n) {
    k_cvt_f32_bf16<<<(unsigned)((n + 255) / 256), 256, 0, stream>>>(in, o, n);
  };

  // 1) precision conversion
  cvt(x,    xb,     M * E_);
  cvt(Wqa,  wqa_b,  (long)QL_ * E_);
  cvt(Wqb,  wqb_b,  (long)(H_ * HD_) * QL_);
  cvt(Wkva, wkva_b, (long)(KVL_ + ROPE_) * E_);
  cvt(Wkvb, wkvb_b, (long)(H_ * (NOPE_ + HD_)) * KVL_);
  cvt(Wout, wout_b, (long)E_ * (H_ * HD_));

  // 2) qc = x @ Wqa^T ; layernorm
  k_gemm_bf16<<<dim3(QL_ / 64, (unsigned)(M / 64)), 256, 0, stream>>>(xb, wqa_b, qc, (int)M, QL_, E_);
  k_layernorm_bf16<<<(unsigned)M, 256, 0, stream>>>(qc, QL_, g_qa, b_qa, qcln);

  // 3) kva = x @ Wkva^T ; layernorm on first 512 cols
  k_gemm_bf16<<<dim3((KVL_ + ROPE_ + 63) / 64, (unsigned)(M / 64)), 256, 0, stream>>>(
      xb, wkva_b, kva, (int)M, KVL_ + ROPE_, E_);
  k_layernorm_bf16<<<(unsigned)M, 256, 0, stream>>>(kva, KVL_ + ROPE_, g_kva, b_kva, ckvln);

  // 4) q = qc_ln @ Wqb^T ; kvd = ckv_ln @ Wkvb^T
  k_gemm_bf16<<<dim3((H_ * HD_) / 64, (unsigned)(M / 64)), 256, 0, stream>>>(
      qcln, wqb_b, qf32, (int)M, H_ * HD_, QL_);
  k_gemm_bf16<<<dim3((H_ * (NOPE_ + HD_)) / 64, (unsigned)(M / 64)), 256, 0, stream>>>(
      ckvln, wkvb_b, kvd, (int)M, H_ * (NOPE_ + HD_), KVL_);

  // 5) RoPE + head-major packing to bf16 (V transposed for contiguous B-fragment loads)
  const unsigned npack = (unsigned)(((long)B_ * H_ * S_ * HD_) / 256);
  k_pack_q<<<npack, 256, 0, stream>>>(qf32, qfull);
  k_pack_kv<<<npack, 256, 0, stream>>>(kvd, kva, kfull, vTb);

  // 6) flash attention (512 WGs x 8 waves, 16 q-rows per wave)
  k_attention<<<(unsigned)(B_ * H_ * (S_ / 128)), 256, 0, stream>>>(qfull, kfull, vTb, attno);

  // 7) final projection: out = attn_out @ Wout^T
  cvt(attno, attno_b, M * (H_ * HD_));
  k_gemm_bf16<<<dim3(E_ / 64, (unsigned)(M / 64)), 256, 0, stream>>>(
      attno_b, wout_b, out, (int)M, E_, H_ * HD_);
}
